// LearningModel_6047313953298
// MI455X (gfx1250) — matrix operand order repaired and checked by reading, not compiled, not run on hardware
//
#include <hip/hip_runtime.h>
#include <hip/hip_bf16.h>
#include <math.h>

// ---------------- problem constants (match reference) ----------------
#define N_CLAUSES 50000
#define FEAT      64
#define HID       256
#define KEYD      256
#define NQ        2
#define T_SEL     1000
#define DISCOUNT  0.995f

typedef __attribute__((ext_vector_type(16))) __bf16 v16bf;
typedef __attribute__((ext_vector_type(8)))  float  v8f;

// ---------------------------------------------------------------------
// K0: A2[q][j] = sum_k W2[j][k]*keysW[q][k];  cfold[q] = keysW[q]·b2
// (folds away the 6.55 GFLOP layer-2 GEMM)
// ---------------------------------------------------------------------
__global__ void fold_keys_kernel(const float* __restrict__ W2,
                                 const float* __restrict__ b2,
                                 const float* __restrict__ keysW,
                                 float* __restrict__ A2,
                                 float* __restrict__ cfold) {
    int idx = blockIdx.x * blockDim.x + threadIdx.x;   // 0..511
    if (idx >= NQ * HID) return;
    int q = idx >> 8, j = idx & (HID - 1);
    float s = 0.f;
    for (int k = 0; k < KEYD; ++k)
        s += W2[(size_t)j * KEYD + k] * keysW[(size_t)q * KEYD + k];
    A2[(size_t)q * HID + j] = s;
    if (j == 0) {
        float c = 0.f;
        for (int k = 0; k < KEYD; ++k)
            c += keysW[(size_t)q * KEYD + k] * b2[k];
        cfold[q] = c;
    }
}

// ---------------------------------------------------------------------
// K1: fused  logits[q][n] = relu(X[n]@W1 + b1) · A2[q] + cfold[q]
// W1 is pre-swizzled into LDS in WMMA B-fragment order so each lane
// reads its 16-element fragment as two aligned ds_load_b128.
//
// Fragment layout (16-bit B 32x16, ISA 7.12.2): lane<16 -> K 0..7,16..23,
// lane>=16 -> K 8..15,24..31 of the 32-K slab; column N = ht*16 + (lane&15).
// LDS index: ((ht*2 + frag)*32 + lane)*16 + p   (bijection over W1)
// ---------------------------------------------------------------------
__global__ __launch_bounds__(128)
void mlp_logits_kernel(const float* __restrict__ feats,
                       const float* __restrict__ W1,
                       const float* __restrict__ b1,
                       const float* __restrict__ A2,
                       const float* __restrict__ cfold,
                       float* __restrict__ logits) {
    __shared__ __bf16 sW1f[FEAT * HID];  // 32 KB, fragment-ordered
    __shared__ float  sB1[HID];

    const int tid = threadIdx.x;
    for (int i = tid; i < FEAT * HID; i += 128) {
        const int p    = i & 15;
        const int ln   = (i >> 4) & 31;
        const int f    = (i >> 9) & 1;
        const int ht   = i >> 10;
        const int K    = f * 32 + ((ln & 16) ? 8 : 0) + (p & 7) + ((p & 8) ? 16 : 0);
        const int col  = ht * 16 + (ln & 15);
        sW1f[i] = (__bf16)W1[(size_t)K * HID + col];
    }
    for (int i = tid; i < HID; i += 128) sB1[i] = b1[i];
    __syncthreads();

    const int wave = tid >> 5;
    const int lane = tid & 31;
    const int tile = blockIdx.x * 4 + wave;           // 16 clauses per wave
    if (tile * 16 >= N_CLAUSES) return;               // uniform per wave -> EXEC all-1 at WMMA

    const int m16    = lane & 15;
    const int row    = tile * 16 + m16;
    const int khalfA = (lane < 16) ? 0 : 8;           // 16-bit A 16x32 layout (ISA 7.12.2)

    // A fragments via guaranteed b128 loads (feature rows are 256B aligned)
    const float4* fr4 = (const float4*)(feats + (size_t)row * FEAT);
    const int q0 = khalfA >> 2;                       // 0 or 2
    const float4 f0 = fr4[q0],     f1 = fr4[q0 + 1];  // K = khalfA..khalfA+7
    const float4 f2 = fr4[q0 + 4], f3 = fr4[q0 + 5];  // K = 16+khalfA..+7
    const float4 f4 = fr4[q0 + 8], f5 = fr4[q0 + 9];  // K = 32+khalfA..+7
    const float4 f6 = fr4[q0 +12], f7 = fr4[q0 +13];  // K = 48+khalfA..+7

    v16bf a0, a1;
    {
        const float c0[16] = {f0.x,f0.y,f0.z,f0.w, f1.x,f1.y,f1.z,f1.w,
                              f2.x,f2.y,f2.z,f2.w, f3.x,f3.y,f3.z,f3.w};
        const float c1[16] = {f4.x,f4.y,f4.z,f4.w, f5.x,f5.y,f5.z,f5.w,
                              f6.x,f6.y,f6.z,f6.w, f7.x,f7.y,f7.z,f7.w};
#pragma unroll
        for (int p = 0; p < 16; ++p) { a0[p] = (__bf16)c0[p]; a1[p] = (__bf16)c1[p]; }
    }

    float acc0[8] = {0.f}, acc1[8] = {0.f};

    for (int ht = 0; ht < HID / 16; ++ht) {
        const int col = ht * 16 + m16;                // hidden unit (B/C column N)
        const v16bf* bp =
            (const v16bf*)&sW1f[(size_t)((ht * 2 + 0) * 32 + lane) * 16];
        const v16bf b0v = bp[0];                      // frag 0: K 0..31  (2x ds_load_b128)
        const v16bf b1v = bp[32];                     // frag 1: K 32..63 (stride 32 lanes)

        v8f c = {};
        c = __builtin_amdgcn_wmma_f32_16x16x32_bf16(false, a0, false, b0v, (short)0, c, false, false);
        c = __builtin_amdgcn_wmma_f32_16x16x32_bf16(false, a1, false, b1v, (short)0, c, false, false);

        const float bias = sB1[col];
        const float a2q0 = A2[col];
        const float a2q1 = A2[HID + col];
#pragma unroll
        for (int r = 0; r < 8; ++r) {
            float h = c[r] + bias;
            h = h > 0.f ? h : 0.f;                    // relu
            acc0[r] += h * a2q0;
            acc1[r] += h * a2q1;
        }
    }

    // Reduce over the 16 lanes of each half (N dim); xor offsets <16 stay in-half.
#pragma unroll
    for (int r = 0; r < 8; ++r) {
#pragma unroll
        for (int off = 8; off >= 1; off >>= 1) {
            acc0[r] += __shfl_xor(acc0[r], off, 32);
            acc1[r] += __shfl_xor(acc1[r], off, 32);
        }
    }

    const float c0 = cfold[0], c1 = cfold[1];
    const int base = tile * 16;
#pragma unroll
    for (int r = 0; r < 8; ++r) {
        if (m16 == r) {                               // lane r -> M=r ; lane r+16 -> M=r+8
            const int m = r + ((lane < 16) ? 0 : 8);
            logits[base + m]             = acc0[r] + c0;
            logits[N_CLAUSES + base + m] = acc1[r] + c1;
        }
    }
}

// ---------------------------------------------------------------------
// K2: one block per SEL event — single pass, 16-byte mask loads.
// Counts via byte-sum trick (bool bytes are 0/1); exp only on passive.
// ev[t] = {num_pass, num_good, ce, 0}
// ---------------------------------------------------------------------
__global__ __launch_bounds__(256)
void event_reduce_kernel(const float* __restrict__ logits,
                         const unsigned char* __restrict__ passive,
                         const unsigned char* __restrict__ good,
                         const int* __restrict__ qidx,
                         float* __restrict__ ev) {
    const int t = blockIdx.x;
    const int q = qidx[t] & 1;
    // rows are 50000 bytes = 16*3125 -> every row is 16B aligned
    const uint4* pm4 = (const uint4*)(passive + (size_t)t * N_CLAUSES);
    const uint4* gm4 = (const uint4*)(good    + (size_t)t * N_CLAUSES);
    const float* lg  = logits + (size_t)q * N_CLAUSES;

    float npass = 0.f, ngood = 0.f, gsum = 0.f;
    float m = -__builtin_inff(), s = 0.f;

    for (int g = threadIdx.x; g < N_CLAUSES / 16; g += 256) {
        __builtin_prefetch(pm4 + g + 512, 0, 1);      // global_prefetch_b8 on the streams
        __builtin_prefetch(gm4 + g + 512, 0, 1);
        const uint4 pw = pm4[g];
        const uint4 gw = gm4[g];

        // byte-wise sums: bytes are 0/1, per-byte totals <=4, grand total <=16
        const unsigned psum = (pw.x + pw.y) + (pw.z + pw.w);
        const unsigned gsum8 = (gw.x + gw.y) + (gw.z + gw.w);
        npass += (float)((psum  * 0x01010101u) >> 24);
        ngood += (float)((gsum8 * 0x01010101u) >> 24);

        const unsigned pws[4] = {pw.x, pw.y, pw.z, pw.w};
        const unsigned gws[4] = {gw.x, gw.y, gw.z, gw.w};
        const int base = g * 16;
#pragma unroll
        for (int wI = 0; wI < 4; ++wI) {
            const unsigned w  = pws[wI];
            if (w == 0u) continue;                    // good subset of passive
            const unsigned gg = gws[wI];
#pragma unroll
            for (int b = 0; b < 4; ++b) {
                if ((w >> (8 * b)) & 0xFFu) {
                    const float x = lg[base + wI * 4 + b];  // logits L2-resident (400 KB)
                    if (x > m) { s *= __expf(m - x); m = x; }
                    s += __expf(x - m);
                    if ((gg >> (8 * b)) & 0xFFu) gsum += x;
                }
            }
        }
    }

    // wave32 reduce of (m,s) + plain sums
#pragma unroll
    for (int off = 16; off >= 1; off >>= 1) {
        const float mo = __shfl_xor(m, off, 32);
        const float so = __shfl_xor(s, off, 32);
        const float mn = fmaxf(m, mo);
        s = (mn == -__builtin_inff()) ? 0.f
                                      : (s * __expf(m - mn) + so * __expf(mo - mn));
        m = mn;
        npass += __shfl_xor(npass, off, 32);
        ngood += __shfl_xor(ngood, off, 32);
        gsum  += __shfl_xor(gsum,  off, 32);
    }

    __shared__ float sm[8], ss[8], sp[8], sg[8], sgs[8];
    const int wv = threadIdx.x >> 5, ln = threadIdx.x & 31;
    if (ln == 0) { sm[wv] = m; ss[wv] = s; sp[wv] = npass; sg[wv] = ngood; sgs[wv] = gsum; }
    __syncthreads();

    if (threadIdx.x == 0) {
        m = sm[0]; s = ss[0]; npass = sp[0]; ngood = sg[0]; gsum = sgs[0];
        for (int w = 1; w < 8; ++w) {
            const float mo = sm[w], so = ss[w];
            const float mn = fmaxf(m, mo);
            s = (mn == -__builtin_inff()) ? 0.f
                                          : (s * __expf(m - mn) + so * __expf(mo - mn));
            m = mn;
            npass += sp[w]; ngood += sg[w]; gsum += sgs[w];
        }
        const float lse = m + __logf(s);
        const float ce  = -(gsum - ngood * lse) / fmaxf(ngood, 1.f);
        ev[t * 4 + 0] = npass;
        ev[t * 4 + 1] = ngood;
        ev[t * 4 + 2] = ce;          // NaN when npass==0, but then valid==false in K3
        ev[t * 4 + 3] = 0.f;
    }
}

// ---------------------------------------------------------------------
// K3: sequential discount-factor combine (inherently serial, 1000 steps)
// ---------------------------------------------------------------------
__global__ void finalize_kernel(const float* __restrict__ ev, float* __restrict__ out) {
    if (threadIdx.x != 0 || blockIdx.x != 0) return;
    float loss = 0.f, factor = 1.f, steps = 0.f;
    for (int t = 0; t < T_SEL; ++t) {
        const float npass = ev[t * 4 + 0];
        const float ngood = ev[t * 4 + 1];
        const float ce    = ev[t * 4 + 2];
        const float nbad  = npass - ngood;
        if (ngood > 0.f && nbad > 0.f) {
            const float w = factor * (nbad / fmaxf(npass, 1.f));
            loss  += w * ce;
            factor *= DISCOUNT;
            steps += 1.f;
        }
    }
    out[0] = loss / fmaxf(steps, 1.f);
}

// ---------------------------------------------------------------------
extern "C" void kernel_launch(void* const* d_in, const int* in_sizes, int n_in,
                              void* d_out, int out_size, void* d_ws, size_t ws_size,
                              hipStream_t stream) {
    // setup_inputs order:
    const float* feats   = (const float*)d_in[0];   // [50000,64]
    const float* W1      = (const float*)d_in[1];   // [64,256]
    const float* b1      = (const float*)d_in[2];   // [256]
    const float* W2      = (const float*)d_in[3];   // [256,256]
    const float* b2      = (const float*)d_in[4];   // [256]
    const float* keysW   = (const float*)d_in[5];   // [2,256]
    const unsigned char* passive = (const unsigned char*)d_in[6]; // [1000,50000] bool
    const unsigned char* good    = (const unsigned char*)d_in[7]; // [1000,50000] bool
    const int* qidx      = (const int*)d_in[8];     // [1000]

    float* ws     = (float*)d_ws;
    float* A2     = ws;                  // 512 floats
    float* cfold  = ws + 512;            // 2 floats (padded to 32)
    float* logits = ws + 544;            // 2*50000 floats
    float* ev     = ws + 544 + 2 * N_CLAUSES; // 4*1000 floats  (~418 KB total)

    fold_keys_kernel<<<2, 256, 0, stream>>>(W2, b2, keysW, A2, cfold);

    const int tiles  = N_CLAUSES / 16;          // 3125
    const int blocks = (tiles + 3) / 4;         // 4 waves/block
    mlp_logits_kernel<<<blocks, 128, 0, stream>>>(feats, W1, b1, A2, cfold, logits);

    event_reduce_kernel<<<T_SEL, 256, 0, stream>>>(logits, passive, good, qidx, ev);

    finalize_kernel<<<1, 1, 0, stream>>>(ev, (float*)d_out);
}
